// AttentionSubsample_17635135717806
// MI455X (gfx1250) — compile-verified, matrix operand order checked by reading, not run on hardware
//
#include <hip/hip_runtime.h>
#include <hip/hip_bf16.h>

typedef __attribute__((ext_vector_type(16))) __bf16 v16bf;
typedef __attribute__((ext_vector_type(8)))  float  v8f;

// ---- problem constants ----
#define BATCH   256
#define NTOK    196     // 14*14
#define NQ      49      // 7*7
#define INDIM   384
#define NHEAD   12
#define KDIM    32
#define VDIM    64
#define DH      768     // NHEAD*VDIM
#define ODIM    512
#define KVH     1152    // NHEAD*(KDIM+VDIM)
#define NPAD    224     // 196 padded to 7*32
#define SCALE   0.17677669529663687f   // 1/sqrt(32)

// LDS byte offset of a shared-memory pointer (low 32 bits of the generic address).
__device__ inline unsigned lds_off(const void* p) {
  return (unsigned)(unsigned long long)p;
}

// CDNA5 async copy: 16 bytes global -> LDS per lane, GVS addressing
// (SGPR base + 32-bit per-lane byte offset), LDS dest address in VGPR.
__device__ inline void async_cp16(unsigned lds_addr, const void* sbase, unsigned gbyteoff) {
  asm volatile("global_load_async_to_lds_b128 %0, %1, %2"
               :: "v"(lds_addr), "v"(gbyteoff), "s"(sbase) : "memory");
}
__device__ inline void async_wait() {
  asm volatile("s_wait_asynccnt 0x0" ::: "memory");
}

// Load a 16x32 bf16 fragment (A or B^T tile) from LDS, row-major, row stride `ld`.
__device__ inline v16bf load_frag(const __bf16* base, int ld, int lane) {
  int m = lane & 15;
  int half = lane >> 4;
  const __bf16* p = base + m * ld + 8 * half;
  v16bf r;
  ((uint4*)&r)[0] = *(const uint4*)p;
  ((uint4*)&r)[1] = *(const uint4*)(p + 16);
  return r;
}

__device__ inline v8f wmma_bf16(v16bf a, v16bf b, v8f c) {
  return __builtin_amdgcn_wmma_f32_16x16x32_bf16(false, a, false, b, (short)0, c, false, false);
}

// ---- prep: fold BN scale into weights (f32 -> bf16) and build bias vector ----
__global__ void fold_kernel(const float* __restrict__ w, const float* __restrict__ g,
                            const float* __restrict__ be, const float* __restrict__ mu,
                            const float* __restrict__ va, int O, int K,
                            __bf16* __restrict__ wbf, float* __restrict__ bout) {
  long i = (long)blockIdx.x * 256 + threadIdx.x;
  long total = (long)O * K;
  if (i < total) {
    long o = i / K;
    float s = g[o] * rsqrtf(va[o] + 1e-5f);
    wbf[i] = (__bf16)(w[i] * s);
  }
  if (i < O) {
    float s = g[i] * rsqrtf(va[i] + 1e-5f);
    bout[i] = be[i] - mu[i] * s;
  }
}

// ---- prep: expand attention bias table to [H][NQ][NTOK] ----
__global__ void biasexp_kernel(const float* __restrict__ ab, const int* __restrict__ idxs,
                               int n_off, float* __restrict__ bias_full) {
  long i = (long)blockIdx.x * 256 + threadIdx.x;
  if (i < (long)NHEAD * NQ * NTOK) {
    long h  = i / (NQ * NTOK);
    long qk = i % (NQ * NTOK);
    bias_full[i] = ab[h * n_off + idxs[qk]];
  }
}

// ---- generic bf16 GEMM: C(64x64 tile) = A(M x K) * W^T, W row-major (O x K) bf16 ----
// AMODE: 0 = A is f32 (x->kv), 1 = f32 subsampled rows (x->q), 2 = bf16 (hidden->proj, async)
// OMODE: 0 = scatter k + transposed v (bf16), 1 = scatter q (bf16), 2 = f32 out stride 512
template <int AMODE, int OMODE>
__global__ __launch_bounds__(128) void gemm_kernel(
    const void* __restrict__ Asrc, const __bf16* __restrict__ Bw,
    const float* __restrict__ bias, int Kdim,
    void* __restrict__ out0, void* __restrict__ out1) {
  __shared__ __bf16 a_sh[64 * 32];
  __shared__ __bf16 b_sh[64 * 32];

  const int tid  = threadIdx.x;
  const int wave = tid >> 5;
  const int lane = tid & 31;
  const int bm = blockIdx.x, bn = blockIdx.y;
  const int half = lane >> 4, lm = lane & 15;

  v8f acc[4];
#pragma unroll
  for (int nt = 0; nt < 4; ++nt)
#pragma unroll
    for (int r = 0; r < 8; ++r) acc[nt][r] = 0.f;

  for (int kk = 0; kk < Kdim; kk += 32) {
    // --- stage B^T tile via async DMA (weights already bf16) ---
#pragma unroll
    for (int i = 0; i < 2; ++i) {
      int e = tid + i * 128;                        // 256 16B chunks
      int r = e >> 2, cc = e & 3;
      unsigned go = (unsigned)(bn * 64 + r);
      async_cp16(lds_off(b_sh + e * 8), Bw, (go * (unsigned)Kdim + kk + cc * 8) * 2u);
    }
    // --- stage A tile (64 rows x 32 K) ---
    if (AMODE == 2) {
      const __bf16* A = (const __bf16*)Asrc;
#pragma unroll
      for (int i = 0; i < 2; ++i) {
        int e = tid + i * 128;
        int r = e >> 2, cc = e & 3;
        unsigned gr = (unsigned)(bm * 64 + r);
        async_cp16(lds_off(a_sh + e * 8), A, (gr * (unsigned)Kdim + kk + cc * 8) * 2u);
      }
    } else {
      const float* A = (const float*)Asrc;
#pragma unroll
      for (int i = 0; i < 4; ++i) {                 // 512 float4 chunks, f32 -> bf16
        int e = tid + i * 128;
        int r = e >> 3, c4 = e & 7;
        long gr = (long)bm * 64 + r;
        if (AMODE == 1) {                           // subsample: row of xs -> row of x
          long b = gr / NQ, np = gr % NQ;
          gr = b * NTOK + (np / 7) * 28 + (np % 7) * 2;
        }
        float4 v = *(const float4*)(A + gr * (long)Kdim + kk + c4 * 4);
        __bf16* d = a_sh + r * 32 + c4 * 4;
        d[0] = (__bf16)v.x; d[1] = (__bf16)v.y; d[2] = (__bf16)v.z; d[3] = (__bf16)v.w;
      }
    }
    async_wait();
    __syncthreads();

    v16bf af = load_frag(a_sh + wave * 16 * 32, 32, lane);
#pragma unroll
    for (int nt = 0; nt < 4; ++nt) {
      v16bf bf = load_frag(b_sh + nt * 16 * 32, 32, lane);
      acc[nt] = wmma_bf16(af, bf, acc[nt]);
    }
    __syncthreads();
  }

  // --- epilogue: hoisted bias, then scatter ---
  float bsv[4];
#pragma unroll
  for (int nt = 0; nt < 4; ++nt) bsv[nt] = bias[bn * 64 + nt * 16 + lm];

#pragma unroll
  for (int nt = 0; nt < 4; ++nt) {
#pragma unroll
    for (int r = 0; r < 8; ++r) {
      int ml = wave * 16 + half * 8 + r;
      int nl = nt * 16 + lm;
      long gr = (long)bm * 64 + ml;
      long gc = (long)bn * 64 + nl;
      float val = acc[nt][r] + bsv[nt];
      if (OMODE == 0) {
        long b = gr / NTOK, n = gr % NTOK;
        long h = gc / 96, rr = gc % 96;
        long bh = b * NHEAD + h;
        if (rr < KDIM)
          ((__bf16*)out0)[(bh * NTOK + n) * KDIM + rr] = (__bf16)val;            // k[bh][n][kd]
        else
          ((__bf16*)out1)[(bh * VDIM + (rr - KDIM)) * NTOK + n] = (__bf16)val;   // vT[bh][d][n]
      } else if (OMODE == 1) {
        long b = gr / NQ, np = gr % NQ;
        long h = gc / KDIM, d = gc % KDIM;
        ((__bf16*)out0)[((b * NHEAD + h) * NQ + np) * KDIM + d] = (__bf16)val;   // q[bh][nq][kd]
      } else {
        ((float*)out0)[gr * ODIM + gc] = val;
      }
    }
  }
}

// ---- fused attention per (b,h): S = scale*q k^T + bias, softmax, O = P v, hard-swish ----
__global__ __launch_bounds__(128) void attn_kernel(
    const __bf16* __restrict__ qbuf, const __bf16* __restrict__ kbuf,
    const __bf16* __restrict__ vtbuf, const float* __restrict__ bias_full,
    __bf16* __restrict__ hidden) {
  __shared__ __bf16 qs[64 * 32];        //  4 KB
  __shared__ __bf16 ks[NPAD * 32];      // 14 KB
  __shared__ __bf16 vts[64 * NPAD];     // 28 KB  (V transposed: [d][token])
  __shared__ float  ss[64 * NPAD];      // 56 KB  (bias prestage, then scores)
  __shared__ __bf16 ps[64 * NPAD];      // 28 KB  (softmax probs bf16)

  const int tid = threadIdx.x;
  const int bh = blockIdx.x;
  const int b = bh / NHEAD, h = bh % NHEAD;
  const __bf16* qsrc = qbuf  + (long)bh * NQ * KDIM;
  const __bf16* ksrc = kbuf  + (long)bh * NTOK * KDIM;
  const __bf16* vsrc = vtbuf + (long)bh * VDIM * NTOK;
  const int wave = tid >> 5, lane = tid & 31, half = lane >> 4, lm = lane & 15;

  // qs: 256 16B chunks, zero-pad rows >= 49
  for (int e = tid; e < 256; e += 128) {
    int r = e >> 2, cc = e & 3;
    uint4 v = make_uint4(0u, 0u, 0u, 0u);
    if (r < NQ) v = *(const uint4*)(qsrc + r * KDIM + cc * 8);
    ((uint4*)qs)[e] = v;
  }
  // ks: 896 16B chunks, zero-pad rows >= 196
  for (int e = tid; e < 896; e += 128) {
    int r = e >> 2, cc = e & 3;
    uint4 v = make_uint4(0u, 0u, 0u, 0u);
    if (r < NTOK) v = *(const uint4*)(ksrc + r * KDIM + cc * 8);
    ((uint4*)ks)[e] = v;
  }
  // vts: 64 rows x 112 dword chunks (224 bf16), zero-pad cols >= 196
  for (int r = wave; r < 64; r += 4) {
    const uint* src = (const uint*)(vsrc + r * NTOK);   // 98 valid dwords
    uint* dst = (uint*)(vts + r * NPAD);
    for (int c = lane; c < 112; c += 32) dst[c] = (c < 98) ? src[c] : 0u;
  }
  // bias prestage into ss: 64 rows x 56 float4 chunks, zero-padded
  for (int e = tid; e < 64 * 56; e += 128) {
    int m = e / 56, c4 = e % 56;
    float4 v = make_float4(0.f, 0.f, 0.f, 0.f);
    if (m < NQ && c4 < 49)
      v = *(const float4*)(bias_full + ((long)h * NQ + m) * NTOK + c4 * 4);
    ((float4*)ss)[e] = v;
  }
  __syncthreads();

  // ---- S = scale * q k^T + bias (bias already in ss) ----
  v16bf aq = load_frag(qs + wave * 16 * 32, 32, lane);
#pragma unroll
  for (int nt = 0; nt < NPAD / 16; ++nt) {
    v16bf bk = load_frag(ks + nt * 16 * 32, 32, lane);
    v8f c;
#pragma unroll
    for (int r = 0; r < 8; ++r) c[r] = 0.f;
    c = wmma_bf16(aq, bk, c);
#pragma unroll
    for (int r = 0; r < 8; ++r) {
      int m = wave * 16 + half * 8 + r;
      int n = nt * 16 + lm;
      ss[m * NPAD + n] = fmaf(c[r], SCALE, ss[m * NPAD + n]);
    }
  }
  __syncthreads();

  // ---- softmax over 196 keys, write probs as bf16 (padded region zeroed) ----
  if (tid < 64) {
    if (tid < NQ) {
      float* row = ss + tid * NPAD;
      float mx = -1e30f;
      for (int n = 0; n < NTOK; ++n) mx = fmaxf(mx, row[n]);
      float sum = 0.f;
      for (int n = 0; n < NTOK; ++n) { float e = __expf(row[n] - mx); row[n] = e; sum += e; }
      float inv = 1.f / sum;
      for (int n = 0; n < NPAD; ++n)
        ps[tid * NPAD + n] = (n < NTOK) ? (__bf16)(row[n] * inv) : (__bf16)0.f;
    } else {
      for (int n = 0; n < NPAD; ++n) ps[tid * NPAD + n] = (__bf16)0.f;
    }
  }
  __syncthreads();

  // ---- O = P V : K = 224 -> 7 accumulating WMMAs per output tile ----
  v8f acc[4];
#pragma unroll
  for (int nt = 0; nt < 4; ++nt)
#pragma unroll
    for (int r = 0; r < 8; ++r) acc[nt][r] = 0.f;

  for (int kt = 0; kt < NPAD / 32; ++kt) {
    v16bf ap = load_frag(ps + (wave * 16) * NPAD + kt * 32, NPAD, lane);
#pragma unroll
    for (int nt = 0; nt < 4; ++nt) {
      v16bf bv = load_frag(vts + (nt * 16) * NPAD + kt * 32, NPAD, lane);
      acc[nt] = wmma_bf16(ap, bv, acc[nt]);
    }
  }

  // ---- hard-swish + store hidden[b][nq][h*64+d] as bf16 ----
#pragma unroll
  for (int nt = 0; nt < 4; ++nt) {
#pragma unroll
    for (int r = 0; r < 8; ++r) {
      int m = wave * 16 + half * 8 + r;
      if (m < NQ) {
        int d = nt * 16 + lm;
        float x = acc[nt][r];
        float hs = x * fminf(fmaxf(x + 3.f, 0.f), 6.f) * (1.f / 6.f);
        hidden[((long)b * NQ + m) * DH + h * VDIM + d] = (__bf16)hs;
      }
    }
  }
}

extern "C" void kernel_launch(void* const* d_in, const int* in_sizes, int n_in,
                              void* d_out, int out_size, void* d_ws, size_t ws_size,
                              hipStream_t stream) {
  const float* x    = (const float*)d_in[0];
  const float* kv_w = (const float*)d_in[1];
  const float* kv_g = (const float*)d_in[2];
  const float* kv_b = (const float*)d_in[3];
  const float* kv_m = (const float*)d_in[4];
  const float* kv_v = (const float*)d_in[5];
  const float* q_w  = (const float*)d_in[6];
  const float* q_g  = (const float*)d_in[7];
  const float* q_b  = (const float*)d_in[8];
  const float* q_m  = (const float*)d_in[9];
  const float* q_v  = (const float*)d_in[10];
  const float* p_w  = (const float*)d_in[11];
  const float* p_g  = (const float*)d_in[12];
  const float* p_b  = (const float*)d_in[13];
  const float* p_m  = (const float*)d_in[14];
  const float* p_v  = (const float*)d_in[15];
  const float* ab   = (const float*)d_in[16];
  const int*   idxs = (const int*)d_in[17];
  const int n_off = in_sizes[16] / NHEAD;

  char* ws = (char*)d_ws;
  auto alloc = [&](size_t bytes) -> char* {
    char* p = ws;
    ws += (bytes + 255) & ~(size_t)255;
    return p;
  };
  __bf16* kvw_bf  = (__bf16*)alloc((size_t)KVH * INDIM * 2);
  float*  kvb     = (float*) alloc((size_t)KVH * 4);
  __bf16* qw_bf   = (__bf16*)alloc((size_t)384 * INDIM * 2);
  float*  qb      = (float*) alloc((size_t)384 * 4);
  __bf16* pw_bf   = (__bf16*)alloc((size_t)ODIM * DH * 2);
  float*  pb      = (float*) alloc((size_t)ODIM * 4);
  float*  biasf   = (float*) alloc((size_t)NHEAD * NQ * NTOK * 4);
  __bf16* kbuf    = (__bf16*)alloc((size_t)BATCH * NHEAD * NTOK * KDIM * 2);
  __bf16* vtbuf   = (__bf16*)alloc((size_t)BATCH * NHEAD * VDIM * NTOK * 2);
  __bf16* qbuf    = (__bf16*)alloc((size_t)BATCH * NHEAD * NQ * KDIM * 2);
  __bf16* hidden  = (__bf16*)alloc((size_t)BATCH * NQ * DH * 2);

  // prep
  fold_kernel<<<(KVH * INDIM + 255) / 256, 256, 0, stream>>>(kv_w, kv_g, kv_b, kv_m, kv_v, KVH, INDIM, kvw_bf, kvb);
  fold_kernel<<<(384 * INDIM + 255) / 256, 256, 0, stream>>>(q_w, q_g, q_b, q_m, q_v, 384, INDIM, qw_bf, qb);
  fold_kernel<<<(ODIM * DH + 255) / 256, 256, 0, stream>>>(p_w, p_g, p_b, p_m, p_v, ODIM, DH, pw_bf, pb);
  biasexp_kernel<<<(NHEAD * NQ * NTOK + 255) / 256, 256, 0, stream>>>(ab, idxs, n_off, biasf);

  // kv GEMM: (50176 x 384) * (384 x 1152), scatter k / v^T
  dim3 g1(50176 / 64, KVH / 64);
  gemm_kernel<0, 0><<<g1, 128, 0, stream>>>(x, kvw_bf, kvb, INDIM, kbuf, vtbuf);
  // q GEMM: (12544 x 384) * (384 x 384), subsampled rows of x
  dim3 g2(12544 / 64, 384 / 64);
  gemm_kernel<1, 1><<<g2, 128, 0, stream>>>(x, qw_bf, qb, INDIM, qbuf, nullptr);
  // fused attention per (b,h)
  attn_kernel<<<BATCH * NHEAD, 128, 0, stream>>>(qbuf, kbuf, vtbuf, biasf, hidden);
  // proj GEMM: (12544 x 768) * (768 x 512) -> f32 out
  dim3 g4(12544 / 64, ODIM / 64);
  gemm_kernel<2, 2><<<g4, 128, 0, stream>>>(hidden, pw_bf, pb, DH, d_out, nullptr);
}
